// GraphConvolution_49512382988455
// MI455X (gfx1250) — matrix operand order repaired and checked by reading, not compiled, not run on hardware
//
#include <hip/hip_runtime.h>
#include <hip/hip_bf16.h>

// ---------------------------------------------------------------------------
// Problem constants (from reference)
// ---------------------------------------------------------------------------
constexpr int NU = 8192, NV = 8192, IN_DIM = 128, HID = 64, NCLS = 5;
constexpr int KTOT = 8192;          // K of the big GEMMs
constexpr int SPLIT = 4;            // K-split for occupancy
constexpr int KPER = KTOT / SPLIT;  // 2048
constexpr int KSTEP = 64;           // K advanced per LDS tile
constexpr int LDS_STRIDE = 72;      // f16 row stride (64 + 8 pad): 36 dwords -> conflict-free
static_assert(NU == NV, "layout assumes square adj");

typedef __attribute__((ext_vector_type(16))) _Float16 v16h;
typedef __attribute__((ext_vector_type(8)))  _Float16 v8h;
typedef __attribute__((ext_vector_type(8)))  float    v8f;
typedef __attribute__((ext_vector_type(4)))  float    v4f;

union V16 { v16h v; v8h h[2]; };

// ---------------------------------------------------------------------------
// Kernel 1: Wu = sum(u_weight[:r+1]), Wv = sum(v_weight[:r+1])
// ---------------------------------------------------------------------------
__global__ __launch_bounds__(256) void wsum_kernel(
    const float* __restrict__ uw, const float* __restrict__ vw,
    const int* __restrict__ rp, float* __restrict__ Wu, float* __restrict__ Wv) {
  int idx = blockIdx.x * 256 + threadIdx.x;        // 0 .. 2*8192-1
  int cnt = *rp + 1;
  if (cnt < 1) cnt = 1;
  if (cnt > NCLS) cnt = NCLS;
  const float* src = (idx < IN_DIM * HID) ? uw : vw;
  float*       dst = (idx < IN_DIM * HID) ? Wu : Wv;
  int e = idx & (IN_DIM * HID - 1);
  float s = 0.f;
  for (int c = 0; c < cnt; ++c) s += src[(size_t)c * IN_DIM * HID + e];
  dst[e] = s;
}

// ---------------------------------------------------------------------------
// Kernel 2: suT[h][i] = (f16)(u[i] . Wu[:,h]);  svT likewise (blockIdx.y)
// ---------------------------------------------------------------------------
__global__ __launch_bounds__(256) void proj_kernel(
    const float* __restrict__ u, const float* __restrict__ v,
    const float* __restrict__ Wu, const float* __restrict__ Wv,
    _Float16* __restrict__ suT, _Float16* __restrict__ svT) {
  __shared__ float Ws[IN_DIM * HID];  // 32 KB
  const float* X = blockIdx.y ? v : u;
  const float* W = blockIdx.y ? Wv : Wu;
  _Float16*  out = blockIdx.y ? svT : suT;

  const int tid = threadIdx.x;
  for (int i = tid; i < IN_DIM * HID; i += 256) Ws[i] = W[i];
  __syncthreads();

  const int row = blockIdx.x * 64 + (tid & 63);
  const int h0  = (tid >> 6) * 16;
  float acc[16];
#pragma unroll
  for (int j = 0; j < 16; ++j) acc[j] = 0.f;

  const float* xrow = X + (size_t)row * IN_DIM;
  for (int k = 0; k < IN_DIM; ++k) {
    float xv = xrow[k];
    const float* wr = Ws + k * HID + h0;
#pragma unroll
    for (int j = 0; j < 16; ++j) acc[j] += xv * wr[j];
  }
#pragma unroll
  for (int j = 0; j < 16; ++j)
    out[(size_t)(h0 + j) * KTOT + row] = (_Float16)acc[j];
}

// ---------------------------------------------------------------------------
// Kernel 3: big WMMA GEMM, double-buffered LDS A tile (128 x 64 f16).
//   z==0 : top = adj   @ sv      (A = adj row-major)
//   z==1 : bot = adj^T @ su      (transpose during LDS store; global coalesced)
// ---------------------------------------------------------------------------
struct ARegs { v4f f[8]; };

__device__ __forceinline__ ARegs load_a_tile(const float* __restrict__ adj,
                                             bool trans, int m0, int kk, int tid,
                                             bool doPrefetch) {
  const float* src;
  if (!trans) {
    const int mr = tid >> 1;              // 0..127
    const int kc = (tid & 1) * 32;        // 0 | 32
    src = adj + (size_t)(m0 + mr) * KTOT + kk + kc;
    if (doPrefetch) __builtin_prefetch(src + 2 * KSTEP, 0, 3);
  } else {
    const int ir = tid >> 2;              // 0..63  (i within K chunk)
    const int jc = (tid & 3) * 32;        // 0..96  (j within M tile)
    src = adj + (size_t)(kk + ir) * KTOT + m0 + jc;
    if (doPrefetch) __builtin_prefetch(src + (size_t)(2 * KSTEP) * KTOT, 0, 3);
  }
  ARegs r;
#pragma unroll
  for (int i = 0; i < 8; ++i) r.f[i] = *(const v4f*)(src + 4 * i);
  return r;
}

__device__ __forceinline__ void store_a_tile(_Float16* __restrict__ buf,
                                             bool trans, int tid, const ARegs& r) {
  if (!trans) {
    const int mr = tid >> 1;
    const int kc = (tid & 1) * 32;
    _Float16* dst = buf + mr * LDS_STRIDE + kc;
#pragma unroll
    for (int j = 0; j < 4; ++j) {
      v8h h;
#pragma unroll
      for (int e = 0; e < 4; ++e) {
        h[e]     = (_Float16)r.f[2 * j][e];
        h[e + 4] = (_Float16)r.f[2 * j + 1][e];
      }
      *(v8h*)(dst + 8 * j) = h;
    }
  } else {
    const int ir = tid >> 2;
    const int jc = (tid & 3) * 32;
    _Float16* dcol = buf + ir;
#pragma unroll
    for (int i = 0; i < 8; ++i)
#pragma unroll
      for (int e = 0; e < 4; ++e)
        dcol[(jc + 4 * i + e) * LDS_STRIDE] = (_Float16)r.f[i][e];
  }
}

__global__ __launch_bounds__(256) void big_gemm_kernel(
    const float* __restrict__ adj,
    const _Float16* __restrict__ suT, const _Float16* __restrict__ svT,
    float* __restrict__ part) {
  __shared__ __align__(16) _Float16 Atile[2][128 * LDS_STRIDE];  // 2 x 18 KB

  const int tid   = threadIdx.x;
  const int lane  = tid & 31;
  const int laneM = lane & 15;
  const int khalf = lane >> 4;
  const int wave  = tid >> 5;

  const int m0 = blockIdx.x * 128;          // output row tile
  const int k0 = blockIdx.y * KPER;         // K-split base
  const bool trans = (blockIdx.z != 0);
  const _Float16* B = trans ? suT : svT;    // B[n][k] f16, [64][8192]
  const int outBase = trans ? NU : 0;

  v8f acc[4] = {};

  // prologue: stage first tile
  ARegs stage = load_a_tile(adj, trans, m0, k0, tid, true);
  store_a_tile(Atile[0], trans, tid, stage);
  __syncthreads();

  constexpr int NIT = KPER / KSTEP;   // 32
  for (int it = 0; it < NIT; ++it) {
    const int kk = k0 + it * KSTEP;

    // issue next tile's global loads early (covered by the WMMA phase)
    if (it + 1 < NIT)
      stage = load_a_tile(adj, trans, m0, kk + KSTEP, tid, it + 2 < NIT);

    const _Float16* curBuf = Atile[it & 1];
#pragma unroll
    for (int s = 0; s < 2; ++s) {       // two K=32 sub-steps per tile
      // A fragment: e<8 -> K = 8*khalf+e ; e>=8 -> K = 16+8*khalf+(e-8)
      V16 a;
      const _Float16* arow = curBuf + (wave * 16 + laneM) * LDS_STRIDE + s * 32;
      a.h[0] = *(const v8h*)(arow + khalf * 8);
      a.h[1] = *(const v8h*)(arow + 16 + khalf * 8);

      // B fragments: e<8 -> K = 8*khalf+e ; e>=8 -> K = 16+8*khalf+(e-8)
      V16 b[4];
#pragma unroll
      for (int t = 0; t < 4; ++t) {
        const _Float16* bp =
            B + (size_t)(16 * t + laneM) * KTOT + kk + s * 32 + khalf * 8;
        b[t].h[0] = *(const v8h*)(bp);
        b[t].h[1] = *(const v8h*)(bp + 16);
      }
#pragma unroll
      for (int t = 0; t < 4; ++t)
        acc[t] = __builtin_amdgcn_wmma_f32_16x16x32_f16(
            false, a.v, false, b[t].v, (short)0, acc[t], false, false);
    }

    if (it + 1 < NIT) store_a_tile(Atile[(it + 1) & 1], trans, tid, stage);
    __syncthreads();
  }

  // write f32 partials: D layout: VGPR v -> row m = v + 8*khalf, col n = 16*t+laneM
  const size_t rowStart = (size_t)blockIdx.y * (2 * NU) + outBase + m0 + wave * 16;
#pragma unroll
  for (int v = 0; v < 8; ++v) {
    float* rp = part + (rowStart + v + 8 * khalf) * HID + laneM;
#pragma unroll
    for (int t = 0; t < 4; ++t) rp[16 * t] = acc[t][v];
  }
}

// ---------------------------------------------------------------------------
// Kernel 4: out[row][h] = degree[row] * sum_split(part) + bias[h]
// ---------------------------------------------------------------------------
__global__ __launch_bounds__(256) void reduce_kernel(
    const float* __restrict__ part, const float* __restrict__ degree,
    const float* __restrict__ bias, float* __restrict__ out) {
  int idx = blockIdx.x * 256 + threadIdx.x;   // 0 .. 16384*64-1
  int row = idx >> 6;
  int h   = idx & 63;
  float s = 0.f;
#pragma unroll
  for (int sp = 0; sp < SPLIT; ++sp)
    s += part[(size_t)sp * (2 * NU) * HID + idx];
  out[idx] = degree[row] * s + bias[h];
}

// ---------------------------------------------------------------------------
extern "C" void kernel_launch(void* const* d_in, const int* in_sizes, int n_in,
                              void* d_out, int out_size, void* d_ws, size_t ws_size,
                              hipStream_t stream) {
  const float* u        = (const float*)d_in[0];
  const float* v        = (const float*)d_in[1];
  const float* adj      = (const float*)d_in[2];
  const float* degree   = (const float*)d_in[3];
  const float* u_weight = (const float*)d_in[4];
  const float* v_weight = (const float*)d_in[5];
  const float* bias     = (const float*)d_in[6];
  const int*   rp       = (const int*)d_in[7];

  // workspace layout
  float*    Wu  = (float*)d_ws;                       // 128*64 f32
  float*    Wv  = Wu + IN_DIM * HID;                  // 128*64 f32
  _Float16* suT = (_Float16*)(Wv + IN_DIM * HID);     // 64*8192 f16
  _Float16* svT = suT + (size_t)HID * KTOT;           // 64*8192 f16
  float*    part = (float*)(svT + (size_t)HID * KTOT);// SPLIT*16384*64 f32

  wsum_kernel<<<(2 * IN_DIM * HID) / 256, 256, 0, stream>>>(u_weight, v_weight, rp, Wu, Wv);
  proj_kernel<<<dim3(NU / 64, 2), 256, 0, stream>>>(u, v, Wu, Wv, suT, svT);
  big_gemm_kernel<<<dim3(NU / 128, SPLIT, 2), 256, 0, stream>>>(adj, suT, svT, part);
  reduce_kernel<<<(2 * NU * HID) / 256, 256, 0, stream>>>(part, degree, bias, (float*)d_out);
}
